// PrecisionRecallGAT_32409823216244
// MI455X (gfx1250) — matrix operand order repaired and checked by reading, not compiled, not run on hardware
//
#include <hip/hip_runtime.h>
#include <cmath>

// ---------------------------------------------------------------------------
// Types for CDNA5 WMMA
// ---------------------------------------------------------------------------
typedef __attribute__((ext_vector_type(16))) _Float16 v16h;
typedef __attribute__((ext_vector_type(8)))  float    v8f;

union Frag16 {
    v16h v;
    unsigned int u[8];
};

#define BM 64
#define BN 32
#define SA 296   // LDS stride (halfs) for A tile rows (>= Kpad max 288, even)
#define SW 296   // LDS stride (halfs) for transposed weight rows

// ---------------------------------------------------------------------------
// Generic GEMM: C[M,Ntot] = A[M,K] @ W[K,Ntot] (+ bias), f16 WMMA, f32 accum.
// Block = 256 threads (8 waves), each wave computes one 16x16 tile of a
// 64x32 block tile. Whole K (zero-padded to mult of 32) staged in LDS.
// ---------------------------------------------------------------------------
__global__ __launch_bounds__(256) void wmma_gemm(
    const float* __restrict__ A, const float* __restrict__ W,
    const float* __restrict__ bias, float* __restrict__ C,
    int M, int K, int Ntot)
{
    __shared__ _Float16 As[BM * SA];
    __shared__ _Float16 Ws[BN * SW];   // transposed: [n][k]

    const int tid  = threadIdx.x;
    const int lane = tid & 31;
    const int wave = tid >> 5;
    const int wm   = wave & 3;     // 4 wave-rows  -> 64 rows
    const int wn   = wave >> 2;    // 2 wave-cols  -> 32 cols
    const int m0   = blockIdx.x * BM;
    const int n0   = blockIdx.y * BN;
    const int Kpad = (K + 31) & ~31;

    // Stage A tile (f32 -> f16), zero padded
    for (int idx = tid; idx < BM * Kpad; idx += 256) {
        int r = idx / Kpad, c = idx - r * Kpad;
        int m = m0 + r;
        float v = 0.f;
        if (m < M && c < K) v = A[(long)m * K + c];
        As[r * SA + c] = (_Float16)v;
    }
    // Stage transposed weight slab (f32 -> f16)
    for (int idx = tid; idx < BN * Kpad; idx += 256) {
        int n = idx / Kpad, k = idx - n * Kpad;
        float v = 0.f;
        if (k < K && (n0 + n) < Ntot) v = W[(long)k * Ntot + n0 + n];
        Ws[n * SW + k] = (_Float16)v;
    }
    __syncthreads();

    v8f acc = {};
    const unsigned int* Au = (const unsigned int*)As;
    const unsigned int* Wu = (const unsigned int*)Ws;
    const int am  = wm * 16 + (lane & 15);
    const int akb = (lane >> 4) * 16;      // K sub-block base for A (0 / 16)
    const int bn  = wn * 16 + (lane & 15);
    const int bkb = (lane >> 4) * 16;      // K sub-block base for B (0 / 16)

    for (int kk = 0; kk < Kpad; kk += 32) {
        Frag16 a, b;
#pragma unroll
        for (int p = 0; p < 8; ++p) {
            a.u[p]     = Au[(am * SA + kk + akb + 2 * p) >> 1];
        }
        const int boff = (bn * SW + kk + bkb) >> 1;
#pragma unroll
        for (int p = 0; p < 8; ++p) b.u[p] = Wu[boff + p];
        acc = __builtin_amdgcn_wmma_f32_16x16x32_f16(
            false, a.v, false, b.v, (short)0, acc, false, false);
    }

    const int ng = n0 + wn * 16 + (lane & 15);
    const float bv = (bias != nullptr && ng < Ntot) ? bias[ng] : 0.f;
#pragma unroll
    for (int v = 0; v < 8; ++v) {
        int mg = m0 + wm * 16 + v + ((lane >> 4) * 8);
        if (mg < M && ng < Ntot) C[(long)mg * Ntot + ng] = acc[v] + bv;
    }
}

// ---------------------------------------------------------------------------
// Classifier layer-1 GEMM with gathered A rows:
//   A[e] = [ h1[src[e]][0:96] | h1[dst[e]][0:96] | edge_feat[e][0:48] ]  (K=240)
// ---------------------------------------------------------------------------
__global__ __launch_bounds__(256) void wmma_gemm_cls(
    const float* __restrict__ h1, const float* __restrict__ ef,
    const int* __restrict__ srcI, const int* __restrict__ dstI,
    const float* __restrict__ W, const float* __restrict__ bias,
    float* __restrict__ C, int E, int Ntot)
{
    __shared__ _Float16 As[BM * SA];
    __shared__ _Float16 Ws[BN * SW];

    const int tid  = threadIdx.x;
    const int lane = tid & 31;
    const int wave = tid >> 5;
    const int wm   = wave & 3;
    const int wn   = wave >> 2;
    const int m0   = blockIdx.x * BM;
    const int n0   = blockIdx.y * BN;
    const int K = 240, Kpad = 256;

    for (int idx = tid; idx < BM * Kpad; idx += 256) {
        int r = idx >> 8, c = idx & 255;
        int e = m0 + r;
        float v = 0.f;
        if (e < E && c < K) {
            if (c < 96)       v = h1[(long)srcI[e] * 96 + c];
            else if (c < 192) v = h1[(long)dstI[e] * 96 + (c - 96)];
            else              v = ef[(long)e * 48 + (c - 192)];
        }
        As[r * SA + c] = (_Float16)v;
    }
    for (int idx = tid; idx < BN * Kpad; idx += 256) {
        int n = idx >> 8, k = idx & 255;
        float v = 0.f;
        if (k < K && (n0 + n) < Ntot) v = W[(long)k * Ntot + n0 + n];
        Ws[n * SW + k] = (_Float16)v;
    }
    __syncthreads();

    v8f acc = {};
    const unsigned int* Au = (const unsigned int*)As;
    const unsigned int* Wu = (const unsigned int*)Ws;
    const int am  = wm * 16 + (lane & 15);
    const int akb = (lane >> 4) * 16;
    const int bn  = wn * 16 + (lane & 15);
    const int bkb = (lane >> 4) * 16;

    for (int kk = 0; kk < Kpad; kk += 32) {
        Frag16 a, b;
#pragma unroll
        for (int p = 0; p < 8; ++p) {
            a.u[p]     = Au[(am * SA + kk + akb + 2 * p) >> 1];
        }
        const int boff = (bn * SW + kk + bkb) >> 1;
#pragma unroll
        for (int p = 0; p < 8; ++p) b.u[p] = Wu[boff + p];
        acc = __builtin_amdgcn_wmma_f32_16x16x32_f16(
            false, a.v, false, b.v, (short)0, acc, false, false);
    }

    const int ng = n0 + wn * 16 + (lane & 15);
    const float bv = (ng < Ntot) ? bias[ng] : 0.f;
#pragma unroll
    for (int v = 0; v < 8; ++v) {
        int mg = m0 + wm * 16 + v + ((lane >> 4) * 8);
        if (mg < E && ng < Ntot) C[(long)mg * Ntot + ng] = acc[v] + bv;
    }
}

// ---------------------------------------------------------------------------
// Segment softmax helpers
// ---------------------------------------------------------------------------
__device__ inline void atomicMaxFloat(float* addr, float val) {
    // sign-split trick; buffer must be initialized to -inf (0xFF800000)
    if (val >= 0.f) atomicMax((int*)addr, __float_as_int(val));
    else            atomicMin((unsigned int*)addr, __float_as_uint(val));
}

// One wave (32 lanes) per edge; C = 96 channels per head (3 per lane).
// pass 0: segment max of logits into mbuf[d*H+h]
// pass 1: a = exp(logit - max); den += a; acc[d] += a * xl[src]
__global__ void gat_edge_pass(
    const float* __restrict__ xl, const float* __restrict__ xr,
    const int* __restrict__ srcI, const int* __restrict__ dstI,
    const float* __restrict__ att,
    float* __restrict__ mbuf, float* __restrict__ den, float* __restrict__ acc,
    int E, int Nn, int H, int pass)
{
    const int lane = threadIdx.x & 31;
    const int e = blockIdx.x * 8 + (threadIdx.x >> 5);
    const int EN = E + Nn;
    if (e >= EN) return;
    int s, d;
    if (e < E) { s = srcI[e]; d = dstI[e]; }
    else       { s = d = e - E; }
    const int C = 96;
    const int HC = H * C;

    for (int h = 0; h < H; ++h) {
        float part = 0.f;
#pragma unroll
        for (int i = 0; i < 3; ++i) {
            int c = i * 32 + lane;
            float v = xl[(long)s * HC + h * C + c] + xr[(long)d * HC + h * C + c];
            v = v > 0.f ? v : 0.2f * v;                 // leaky_relu(0.2)
            part += att[h * C + c] * v;
        }
        for (int off = 16; off > 0; off >>= 1) part += __shfl_xor(part, off);

        if (pass == 0) {
            if (lane == 0) atomicMaxFloat(&mbuf[(long)d * H + h], part);
        } else {
            float a = __expf(part - mbuf[(long)d * H + h]);
            if (lane == 0) atomicAdd(&den[(long)d * H + h], a);
#pragma unroll
            for (int i = 0; i < 3; ++i) {
                int c = i * 32 + lane;
                atomicAdd(&acc[(long)d * HC + h * C + c],
                          a * xl[(long)s * HC + h * C + c]);
            }
        }
    }
}

// acc[n,f] = acc[n,f] / (den[n, f/C] + 1e-16) + bias[f]
__global__ void gat_fin(float* __restrict__ acc, const float* __restrict__ den,
                        const float* __restrict__ bias, int Nn, int H, int C)
{
    long idx = (long)blockIdx.x * blockDim.x + threadIdx.x;
    long total = (long)Nn * H * C;
    if (idx >= total) return;
    int HC = H * C;
    int f = (int)(idx % HC);
    long node = idx / HC;
    int h = f / C;
    acc[idx] = acc[idx] / (den[node * H + h] + 1e-16f) + bias[f];
}

// ---------------------------------------------------------------------------
// BatchNorm (batch statistics) — two pass
// ---------------------------------------------------------------------------
// blockDim.x == F; each block reduces 128 rows, atomicAdd into stats[0:F]=sum,
// stats[F:2F]=sumsq
__global__ void col_stats(const float* __restrict__ X, float* __restrict__ stats,
                          int M, int F)
{
    int col = threadIdx.x;
    long r0 = (long)blockIdx.x * 128;
    float s = 0.f, q = 0.f;
    for (int r = 0; r < 128; ++r) {
        long row = r0 + r;
        if (row < M) {
            float v = X[row * F + col];
            s += v; q += v * v;
        }
    }
    atomicAdd(&stats[col], s);
    atomicAdd(&stats[F + col], q);
}

// act: 0 = none, 1 = relu, 2 = elu
__global__ void bn_act(float* __restrict__ X, const float* __restrict__ stats,
                       const float* __restrict__ g, const float* __restrict__ b,
                       long M, int F, int act)
{
    long idx = (long)blockIdx.x * blockDim.x + threadIdx.x;
    long total = M * F;
    if (idx >= total) return;
    int col = (int)(idx % F);
    float mean = stats[col] / (float)M;
    float var  = stats[F + col] / (float)M - mean * mean;
    float y = (X[idx] - mean) * rsqrtf(var + 1e-5f) * g[col] + b[col];
    if (act == 1)      y = y > 0.f ? y : 0.f;
    else if (act == 2) y = y > 0.f ? y : (__expf(y) - 1.f);
    X[idx] = y;
}

__global__ void fill_f32(float* __restrict__ p, float v, long n)
{
    long i = (long)blockIdx.x * blockDim.x + threadIdx.x;
    if (i < n) p[i] = v;
}

// Final classifier layer: [E,24] @ [24,2] + b  -> d_out
__global__ void final_fc(const float* __restrict__ X, const float* __restrict__ W,
                         const float* __restrict__ b, float* __restrict__ out, int E)
{
    int e = blockIdx.x * blockDim.x + threadIdx.x;
    if (e >= E) return;
    float o0 = b[0], o1 = b[1];
#pragma unroll
    for (int k = 0; k < 24; ++k) {
        float v = X[(long)e * 24 + k];
        o0 += v * W[k * 2 + 0];
        o1 += v * W[k * 2 + 1];
    }
    out[(long)e * 2 + 0] = o0;
    out[(long)e * 2 + 1] = o1;
}

// ---------------------------------------------------------------------------
// Orchestration
// ---------------------------------------------------------------------------
extern "C" void kernel_launch(void* const* d_in, const int* in_sizes, int n_in,
                              void* d_out, int out_size, void* d_ws, size_t ws_size,
                              hipStream_t stream)
{
    const int E = 500000, Nn = 50000;
    const int EN = E + Nn;

    const float* x         = (const float*)d_in[0];
    const int*   eidx      = (const int*)d_in[1];
    const int*   srcI      = eidx;
    const int*   dstI      = eidx + E;
    const float* edge_attr = (const float*)d_in[2];
    const float* ew1  = (const float*)d_in[3];
    const float* eb1  = (const float*)d_in[4];
    const float* eg1  = (const float*)d_in[5];
    const float* ebt1 = (const float*)d_in[6];
    const float* ew2  = (const float*)d_in[7];
    const float* eb2  = (const float*)d_in[8];
    const float* eg2  = (const float*)d_in[9];
    const float* ebt2 = (const float*)d_in[10];
    const float* g0wl = (const float*)d_in[11];
    const float* g0wr = (const float*)d_in[12];
    const float* g0att= (const float*)d_in[13];
    const float* g0b  = (const float*)d_in[14];
    const float* bn0g = (const float*)d_in[15];
    const float* bn0b = (const float*)d_in[16];
    const float* g1wl = (const float*)d_in[17];
    const float* g1wr = (const float*)d_in[18];
    const float* g1att= (const float*)d_in[19];
    const float* g1b  = (const float*)d_in[20];
    const float* bn1g = (const float*)d_in[21];
    const float* bn1b = (const float*)d_in[22];
    const float* cw1  = (const float*)d_in[23];
    const float* cb1  = (const float*)d_in[24];
    const float* cg1  = (const float*)d_in[25];
    const float* cbt1 = (const float*)d_in[26];
    const float* cw2  = (const float*)d_in[27];
    const float* cb2  = (const float*)d_in[28];
    const float* cg2  = (const float*)d_in[29];
    const float* cbt2 = (const float*)d_in[30];
    const float* cw3  = (const float*)d_in[31];
    const float* cb3  = (const float*)d_in[32];
    const float* cg3  = (const float*)d_in[33];
    const float* cbt3 = (const float*)d_in[34];
    const float* cw4  = (const float*)d_in[35];
    const float* cb4  = (const float*)d_in[36];

    // Workspace bump allocator (256B aligned)
    char* wp = (char*)d_ws;
    auto alloc = [&](size_t bytes) -> void* {
        void* r = (void*)wp;
        wp += (bytes + 255) & ~(size_t)255;
        return r;
    };
    float* z1    = (float*)alloc((size_t)E * 96 * 4);   // edge h / classifier c1
    float* z2    = (float*)alloc((size_t)E * 48 * 4);   // edge_feat / classifier c2
    float* xl0   = (float*)alloc((size_t)Nn * 288 * 4);
    float* xr0   = (float*)alloc((size_t)Nn * 288 * 4);
    float* acc0  = (float*)alloc((size_t)Nn * 288 * 4); // GAT0 numer -> h0
    float* xl1   = (float*)alloc((size_t)Nn * 96 * 4);
    float* xr1   = (float*)alloc((size_t)Nn * 96 * 4);
    float* acc1  = (float*)alloc((size_t)Nn * 96 * 4);  // GAT1 numer -> h1
    float* c3    = (float*)alloc((size_t)E * 24 * 4);
    float* m0    = (float*)alloc((size_t)Nn * 3 * 4);
    float* d0    = (float*)alloc((size_t)Nn * 3 * 4);
    float* m1    = (float*)alloc((size_t)Nn * 4);
    float* d1    = (float*)alloc((size_t)Nn * 4);
    float* stats = (float*)alloc(2 * 288 * 4);

    const dim3 blk(256);
    auto ggrid = [&](int M, int N) { return dim3((M + BM - 1) / BM, (N + BN - 1) / BN); };
    auto lgrid = [](long total) { return dim3((unsigned)((total + 255) / 256)); };
    const dim3 gatGrid((EN + 7) / 8);

    // ---- Edge MLP layer 1: z1 = edge_attr @ ew1 + eb1; BN; ReLU
    hipMemsetAsync(stats, 0, 2 * 288 * 4, stream);
    wmma_gemm<<<ggrid(E, 96), blk, 0, stream>>>(edge_attr, ew1, eb1, z1, E, 32, 96);
    col_stats<<<dim3((E + 127) / 128), dim3(96), 0, stream>>>(z1, stats, E, 96);
    bn_act<<<lgrid((long)E * 96), blk, 0, stream>>>(z1, stats, eg1, ebt1, E, 96, 1);

    // ---- Edge MLP layer 2: z2 = z1 @ ew2 + eb2; BN; ReLU -> edge_feat
    hipMemsetAsync(stats, 0, 2 * 288 * 4, stream);
    wmma_gemm<<<ggrid(E, 48), blk, 0, stream>>>(z1, ew2, eb2, z2, E, 96, 48);
    col_stats<<<dim3((E + 127) / 128), dim3(48), 0, stream>>>(z2, stats, E, 48);
    bn_act<<<lgrid((long)E * 48), blk, 0, stream>>>(z2, stats, eg2, ebt2, E, 48, 1);

    // ---- GAT layer 0 projections (no bias here; GATv2 bias applied post-agg)
    wmma_gemm<<<ggrid(Nn, 288), blk, 0, stream>>>(x, g0wl, nullptr, xl0, Nn, 128, 288);
    wmma_gemm<<<ggrid(Nn, 288), blk, 0, stream>>>(x, g0wr, nullptr, xr0, Nn, 128, 288);

    // ---- GAT layer 0 segment softmax + aggregation
    fill_f32<<<lgrid((long)Nn * 3), blk, 0, stream>>>(m0, -INFINITY, (long)Nn * 3);
    hipMemsetAsync(d0, 0, (size_t)Nn * 3 * 4, stream);
    hipMemsetAsync(acc0, 0, (size_t)Nn * 288 * 4, stream);
    gat_edge_pass<<<gatGrid, blk, 0, stream>>>(xl0, xr0, srcI, dstI, g0att,
                                               m0, d0, acc0, E, Nn, 3, 0);
    gat_edge_pass<<<gatGrid, blk, 0, stream>>>(xl0, xr0, srcI, dstI, g0att,
                                               m0, d0, acc0, E, Nn, 3, 1);
    gat_fin<<<lgrid((long)Nn * 288), blk, 0, stream>>>(acc0, d0, g0b, Nn, 3, 96);
    hipMemsetAsync(stats, 0, 2 * 288 * 4, stream);
    col_stats<<<dim3((Nn + 127) / 128), dim3(288), 0, stream>>>(acc0, stats, Nn, 288);
    bn_act<<<lgrid((long)Nn * 288), blk, 0, stream>>>(acc0, stats, bn0g, bn0b, Nn, 288, 2);

    // ---- GAT layer 1 projections
    wmma_gemm<<<ggrid(Nn, 96), blk, 0, stream>>>(acc0, g1wl, nullptr, xl1, Nn, 288, 96);
    wmma_gemm<<<ggrid(Nn, 96), blk, 0, stream>>>(acc0, g1wr, nullptr, xr1, Nn, 288, 96);

    // ---- GAT layer 1 segment softmax + aggregation (heads=1)
    fill_f32<<<lgrid((long)Nn), blk, 0, stream>>>(m1, -INFINITY, (long)Nn);
    hipMemsetAsync(d1, 0, (size_t)Nn * 4, stream);
    hipMemsetAsync(acc1, 0, (size_t)Nn * 96 * 4, stream);
    gat_edge_pass<<<gatGrid, blk, 0, stream>>>(xl1, xr1, srcI, dstI, g1att,
                                               m1, d1, acc1, E, Nn, 1, 0);
    gat_edge_pass<<<gatGrid, blk, 0, stream>>>(xl1, xr1, srcI, dstI, g1att,
                                               m1, d1, acc1, E, Nn, 1, 1);
    gat_fin<<<lgrid((long)Nn * 96), blk, 0, stream>>>(acc1, d1, g1b, Nn, 1, 96);
    hipMemsetAsync(stats, 0, 2 * 288 * 4, stream);
    col_stats<<<dim3((Nn + 127) / 128), dim3(96), 0, stream>>>(acc1, stats, Nn, 96);
    bn_act<<<lgrid((long)Nn * 96), blk, 0, stream>>>(acc1, stats, bn1g, bn1b, Nn, 96, 2);

    // ---- Classifier L1: gathered [h1[src] | h1[dst] | edge_feat] @ cw1 + cb1
    hipMemsetAsync(stats, 0, 2 * 288 * 4, stream);
    wmma_gemm_cls<<<ggrid(E, 96), blk, 0, stream>>>(acc1, z2, srcI, dstI,
                                                    cw1, cb1, z1, E, 96);
    col_stats<<<dim3((E + 127) / 128), dim3(96), 0, stream>>>(z1, stats, E, 96);
    bn_act<<<lgrid((long)E * 96), blk, 0, stream>>>(z1, stats, cg1, cbt1, E, 96, 1);

    // ---- Classifier L2 (edge_feat dead -> reuse z2)
    hipMemsetAsync(stats, 0, 2 * 288 * 4, stream);
    wmma_gemm<<<ggrid(E, 48), blk, 0, stream>>>(z1, cw2, cb2, z2, E, 96, 48);
    col_stats<<<dim3((E + 127) / 128), dim3(48), 0, stream>>>(z2, stats, E, 48);
    bn_act<<<lgrid((long)E * 48), blk, 0, stream>>>(z2, stats, cg2, cbt2, E, 48, 1);

    // ---- Classifier L3
    hipMemsetAsync(stats, 0, 2 * 288 * 4, stream);
    wmma_gemm<<<ggrid(E, 24), blk, 0, stream>>>(z2, cw3, cb3, c3, E, 48, 24);
    col_stats<<<dim3((E + 127) / 128), dim3(24), 0, stream>>>(c3, stats, E, 24);
    bn_act<<<lgrid((long)E * 24), blk, 0, stream>>>(c3, stats, cg3, cbt3, E, 24, 1);

    // ---- Final linear -> d_out [E, 2]
    final_fc<<<dim3((E + 255) / 256), blk, 0, stream>>>(c3, cw4, cb4,
                                                        (float*)d_out, E);
}
